// DecoderLayer_41601053229607
// MI455X (gfx1250) — matrix operand order repaired and checked by reading, not compiled, not run on hardware
//
#include <hip/hip_runtime.h>

// ---------------------------------------------------------------------------
// Types for CDNA5 WMMA (wave32): v_wmma_f32_16x16x32_bf16
// ---------------------------------------------------------------------------
typedef __attribute__((ext_vector_type(16))) __bf16 bf16x16;
typedef __attribute__((ext_vector_type(8)))  float  f32x8;

union Frag16 {
    bf16x16 v;
    uint4   q[2];   // 32 bytes
};

#define WMMA_BF16(a, b, c) \
    __builtin_amdgcn_wmma_f32_16x16x32_bf16(false, (a), false, (b), (short)0, (c), false, false)

// Problem constants
#define BSZ   4
#define SEQ   1024
#define DIM   1024
#define NH    16
#define DH    64
#define DFF   4096
#define BSROW (BSZ * SEQ)   // 4096

// ---------------------------------------------------------------------------
// fp32 -> bf16 convert
// ---------------------------------------------------------------------------
__global__ void cvt_bf16_kernel(const float* __restrict__ x, __bf16* __restrict__ y, int n) {
    int i = blockIdx.x * blockDim.x + threadIdx.x;
    if (i < n) y[i] = (__bf16)x[i];
}

// fp32 [R x C] -> bf16 [C x R]  (weight transpose so GEMM B-frags are contiguous)
__global__ void transpose_w_kernel(const float* __restrict__ W, __bf16* __restrict__ Wt,
                                   int R, int C) {
    size_t idx = (size_t)blockIdx.x * blockDim.x + threadIdx.x;
    if (idx >= (size_t)R * C) return;
    int r = (int)(idx / C);
    int c = (int)(idx % C);
    Wt[(size_t)c * R + r] = (__bf16)W[idx];
}

// V [B,S,H,DH] bf16 -> Vt [B,H,DH,S] bf16 (so P·V B-frags are contiguous in kpos)
__global__ void transpose_v_kernel(const __bf16* __restrict__ V, __bf16* __restrict__ Vt) {
    size_t idx = (size_t)blockIdx.x * blockDim.x + threadIdx.x;  // over B*S*D
    int dd = (int)(idx & 63);
    int h  = (int)((idx >> 6) & 15);
    int s  = (int)((idx >> 10) & (SEQ - 1));
    int b  = (int)(idx >> 20);
    Vt[(((size_t)(b * NH + h) * DH + dd) * SEQ) + s] = V[idx];
}

// ---------------------------------------------------------------------------
// GEMM: out[M,N] = A[M,K](bf16, row stride lda) @ Bt[N,K]^T + bias (+residual)(relu?)
// One wave -> 32x64 tile: two resident A-fragment rows x 4 B tiles = 8 WMMAs
// per K=32 step; every B-fragment is reused twice (~22 FLOP/B from L2).
// Block = 256 threads = 8 waves stacked in M -> 256x64 block tile.
// ---------------------------------------------------------------------------
__global__ __launch_bounds__(256)
void gemm_bf16_kernel(const __bf16* __restrict__ A, int lda,
                      const __bf16* __restrict__ Bt,
                      const float* __restrict__ bias,
                      const float* __restrict__ residual,
                      float* __restrict__ outF,
                      __bf16* __restrict__ outB,
                      int M, int N, int K, int relu) {
    const int lane = threadIdx.x & 31;
    const int wave = threadIdx.x >> 5;
    const int m0 = (blockIdx.x * 8 + wave) * 32;
    const int n0 = blockIdx.y * 64;
    if (m0 >= M) return;

    const int g   = lane >> 4;    // half-wave group
    const int l16 = lane & 15;

    f32x8 accA0 = {}, accA1 = {}, accA2 = {}, accA3 = {};
    f32x8 accB0 = {}, accB1 = {}, accB2 = {}, accB3 = {};
    const __bf16* arow0 = A + (size_t)(m0 + l16) * lda;        // rows m0..m0+15
    const __bf16* arow1 = A + (size_t)(m0 + 16 + l16) * lda;   // rows m0+16..m0+31

    for (int k0 = 0; k0 < K; k0 += 32) {
        Frag16 af0, af1;
        af0.q[0] = *(const uint4*)(arow0 + k0 + 16 * g);       // k = 16g .. 16g+7
        af0.q[1] = *(const uint4*)(arow0 + k0 + 16 * g + 8);   // k = 16g+8 ..
        af1.q[0] = *(const uint4*)(arow1 + k0 + 16 * g);
        af1.q[1] = *(const uint4*)(arow1 + k0 + 16 * g + 8);

        Frag16 bf0, bf1, bf2, bf3;
        const __bf16* b0 = Bt + (size_t)(n0 + 0 * 16 + l16) * K + k0 + 16 * g;
        const __bf16* b1 = Bt + (size_t)(n0 + 1 * 16 + l16) * K + k0 + 16 * g;
        const __bf16* b2 = Bt + (size_t)(n0 + 2 * 16 + l16) * K + k0 + 16 * g;
        const __bf16* b3 = Bt + (size_t)(n0 + 3 * 16 + l16) * K + k0 + 16 * g;
        bf0.q[0] = *(const uint4*)(b0);  bf0.q[1] = *(const uint4*)(b0 + 8);
        bf1.q[0] = *(const uint4*)(b1);  bf1.q[1] = *(const uint4*)(b1 + 8);
        bf2.q[0] = *(const uint4*)(b2);  bf2.q[1] = *(const uint4*)(b2 + 8);
        bf3.q[0] = *(const uint4*)(b3);  bf3.q[1] = *(const uint4*)(b3 + 8);

        accA0 = WMMA_BF16(af0.v, bf0.v, accA0);
        accB0 = WMMA_BF16(af1.v, bf0.v, accB0);
        accA1 = WMMA_BF16(af0.v, bf1.v, accA1);
        accB1 = WMMA_BF16(af1.v, bf1.v, accB1);
        accA2 = WMMA_BF16(af0.v, bf2.v, accA2);
        accB2 = WMMA_BF16(af1.v, bf2.v, accB2);
        accA3 = WMMA_BF16(af0.v, bf3.v, accA3);
        accB3 = WMMA_BF16(af1.v, bf3.v, accB3);
    }

    f32x8 accs[2][4] = {{accA0, accA1, accA2, accA3}, {accB0, accB1, accB2, accB3}};
#pragma unroll
    for (int half = 0; half < 2; ++half) {
#pragma unroll
        for (int t = 0; t < 4; ++t) {
            const int col = n0 + t * 16 + l16;
            const float bb = bias ? bias[col] : 0.0f;
#pragma unroll
            for (int r = 0; r < 8; ++r) {
                const int row = m0 + half * 16 + r + 8 * g;
                float v = accs[half][t][r] + bb;
                if (residual) v += residual[(size_t)row * N + col];
                if (relu)     v = fmaxf(v, 0.0f);
                if (outF) outF[(size_t)row * N + col] = v;
                if (outB) outB[(size_t)row * N + col] = (__bf16)v;
            }
        }
    }
}

// ---------------------------------------------------------------------------
// Flash-style attention. Grid: (S/64, H, B), block = 128 (4 waves).
// Each wave owns one 16-row q-tile; streams keys 32 at a time with online
// softmax; P bounces through a private LDS tile to re-layout C-frag -> A-frag.
// ---------------------------------------------------------------------------
__global__ __launch_bounds__(128)
void attn_kernel(const __bf16* __restrict__ Q, const __bf16* __restrict__ Kb,
                 const __bf16* __restrict__ Vt, __bf16* __restrict__ Out,
                 int causal) {
    __shared__ __align__(16) __bf16 pbuf[4][16 * 32];

    const int lane = threadIdx.x & 31;
    const int wave = threadIdx.x >> 5;
    const int h = blockIdx.y;
    const int b = blockIdx.z;
    const int qbase = (blockIdx.x * 4 + wave) * 16;
    const int g = lane >> 4;
    const int l16 = lane & 15;
    __bf16* pb = &pbuf[wave][0];

    // Resident Q fragments (d = 64 -> two 16x32 A-frags)
    const __bf16* qrow = Q + (size_t)(b * SEQ + qbase + l16) * DIM + h * DH;
    Frag16 qf0, qf1;
    qf0.q[0] = *(const uint4*)(qrow + 16 * g);
    qf0.q[1] = *(const uint4*)(qrow + 16 * g + 8);
    qf1.q[0] = *(const uint4*)(qrow + 32 + 16 * g);
    qf1.q[1] = *(const uint4*)(qrow + 40 + 16 * g);

    float mrow[8], lrow[8];
    f32x8 o0 = {}, o1 = {}, o2 = {}, o3 = {};
#pragma unroll
    for (int r = 0; r < 8; ++r) { mrow[r] = -INFINITY; lrow[r] = 0.0f; }

    const float scale = 0.125f;  // 1/sqrt(64)
    const int kend = causal ? (qbase + 16) : SEQ;

    for (int kb = 0; kb < kend; kb += 32) {
        // ---- scores: two 16x16 tiles (cols kb..+15, kb+16..+31)
        f32x8 s0 = {}, s1 = {};
        const __bf16* krow0 = Kb + (size_t)(b * SEQ + kb + l16) * DIM + h * DH;
        const __bf16* krow1 = krow0 + (size_t)16 * DIM;
        Frag16 kf;
        kf.q[0] = *(const uint4*)(krow0 + 16 * g);
        kf.q[1] = *(const uint4*)(krow0 + 16 * g + 8);
        s0 = WMMA_BF16(qf0.v, kf.v, s0);
        kf.q[0] = *(const uint4*)(krow0 + 32 + 16 * g);
        kf.q[1] = *(const uint4*)(krow0 + 40 + 16 * g);
        s0 = WMMA_BF16(qf1.v, kf.v, s0);
        kf.q[0] = *(const uint4*)(krow1 + 16 * g);
        kf.q[1] = *(const uint4*)(krow1 + 16 * g + 8);
        s1 = WMMA_BF16(qf0.v, kf.v, s1);
        kf.q[0] = *(const uint4*)(krow1 + 32 + 16 * g);
        kf.q[1] = *(const uint4*)(krow1 + 40 + 16 * g);
        s1 = WMMA_BF16(qf1.v, kf.v, s1);

        // ---- online softmax update (row-wise over 16-lane halves)
#pragma unroll
        for (int r = 0; r < 8; ++r) {
            const int row = qbase + r + 8 * g;
            float a = s0[r] * scale;
            float c = s1[r] * scale;
            if (causal) {
                a = (kb + l16       > row) ? -1e30f : a;
                c = (kb + 16 + l16  > row) ? -1e30f : c;
            }
            float t = fmaxf(a, c);
            t = fmaxf(t, __shfl_xor(t, 1, 32));
            t = fmaxf(t, __shfl_xor(t, 2, 32));
            t = fmaxf(t, __shfl_xor(t, 4, 32));
            t = fmaxf(t, __shfl_xor(t, 8, 32));
            const float mnew = fmaxf(mrow[r], t);
            const float p0 = __expf(a - mnew);
            const float p1 = __expf(c - mnew);
            float ls = p0 + p1;
            ls += __shfl_xor(ls, 1, 32);
            ls += __shfl_xor(ls, 2, 32);
            ls += __shfl_xor(ls, 4, 32);
            ls += __shfl_xor(ls, 8, 32);
            const float corr = __expf(mrow[r] - mnew);
            lrow[r] = lrow[r] * corr + ls;
            mrow[r] = mnew;
            o0[r] *= corr; o1[r] *= corr; o2[r] *= corr; o3[r] *= corr;
            // stash P (C layout row r+8g, cols l16 / l16+16) to LDS row-major
            pb[(r + 8 * g) * 32 + l16]      = (__bf16)p0;
            pb[(r + 8 * g) * 32 + 16 + l16] = (__bf16)p1;
        }
        // same-wave LDS RAW: wait on DScnt (no barrier; waves run independently)
        asm volatile("s_wait_dscnt 0" ::: "memory");

        // ---- reload P as an A fragment (16x32 bf16)
        Frag16 pf;
        pf.q[0] = *(const uint4*)(pb + l16 * 32 + 16 * g);
        pf.q[1] = *(const uint4*)(pb + l16 * 32 + 16 * g + 8);

        // ---- O += P @ V   (Vt is [B,H,DH,S] so B-frags are contiguous)
        const __bf16* vb = Vt + (size_t)((b * NH + h) * DH) * SEQ + kb + 16 * g;
        Frag16 vf;
        vf.q[0] = *(const uint4*)(vb + (size_t)(0 * 16 + l16) * SEQ);
        vf.q[1] = *(const uint4*)(vb + (size_t)(0 * 16 + l16) * SEQ + 8);
        o0 = WMMA_BF16(pf.v, vf.v, o0);
        vf.q[0] = *(const uint4*)(vb + (size_t)(1 * 16 + l16) * SEQ);
        vf.q[1] = *(const uint4*)(vb + (size_t)(1 * 16 + l16) * SEQ + 8);
        o1 = WMMA_BF16(pf.v, vf.v, o1);
        vf.q[0] = *(const uint4*)(vb + (size_t)(2 * 16 + l16) * SEQ);
        vf.q[1] = *(const uint4*)(vb + (size_t)(2 * 16 + l16) * SEQ + 8);
        o2 = WMMA_BF16(pf.v, vf.v, o2);
        vf.q[0] = *(const uint4*)(vb + (size_t)(3 * 16 + l16) * SEQ);
        vf.q[1] = *(const uint4*)(vb + (size_t)(3 * 16 + l16) * SEQ + 8);
        o3 = WMMA_BF16(pf.v, vf.v, o3);
    }

    // ---- normalize and write out (bf16, [B,S,H,DH] row-major)
    f32x8 oacc[4] = {o0, o1, o2, o3};
#pragma unroll
    for (int t = 0; t < 4; ++t) {
#pragma unroll
        for (int r = 0; r < 8; ++r) {
            const float v = oacc[t][r] / lrow[r];
            Out[(size_t)(b * SEQ + qbase + r + 8 * g) * DIM + h * DH + t * 16 + l16] = (__bf16)v;
        }
    }
}

// ---------------------------------------------------------------------------
// Row LayerNorm (eps = 1e-3), dual f32 / bf16 output. One block per row.
// ---------------------------------------------------------------------------
__global__ __launch_bounds__(256)
void ln_kernel(const float* __restrict__ x, const float* __restrict__ gamma,
               const float* __restrict__ beta, float* __restrict__ outF,
               __bf16* __restrict__ outB, int D) {
    __shared__ float red[256];
    const int row = blockIdx.x;
    const int tid = threadIdx.x;
    const float* xr = x + (size_t)row * D;

    float s = 0.0f, s2 = 0.0f;
    for (int i = tid; i < D; i += 256) { const float v = xr[i]; s += v; s2 += v * v; }

    red[tid] = s; __syncthreads();
    for (int st = 128; st > 0; st >>= 1) { if (tid < st) red[tid] += red[tid + st]; __syncthreads(); }
    const float mean = red[0] / (float)D;
    __syncthreads();
    red[tid] = s2; __syncthreads();
    for (int st = 128; st > 0; st >>= 1) { if (tid < st) red[tid] += red[tid + st]; __syncthreads(); }
    const float var = red[0] / (float)D - mean * mean;
    const float rstd = rsqrtf(var + 1e-3f);

    for (int i = tid; i < D; i += 256) {
        const float v = (xr[i] - mean) * rstd * gamma[i] + beta[i];
        if (outF) outF[(size_t)row * D + i] = v;
        if (outB) outB[(size_t)row * D + i] = (__bf16)v;
    }
}

// ---------------------------------------------------------------------------
// Host-side orchestration
// ---------------------------------------------------------------------------
static void launch_gemm(hipStream_t st, const __bf16* A, int lda, const __bf16* Bt,
                        const float* bias, const float* resid,
                        float* outF, __bf16* outB, int M, int N, int K, int relu) {
    dim3 grid(M / 256, N / 64);
    gemm_bf16_kernel<<<grid, 256, 0, st>>>(A, lda, Bt, bias, resid, outF, outB, M, N, K, relu);
}

extern "C" void kernel_launch(void* const* d_in, const int* in_sizes, int n_in,
                              void* d_out, int out_size, void* d_ws, size_t ws_size,
                              hipStream_t stream) {
    // setup_inputs() dict order
    const float* inputs = (const float*)d_in[0];
    const float* enc    = (const float*)d_in[1];
    const float* wq1 = (const float*)d_in[2];
    const float* wk1 = (const float*)d_in[3];
    const float* wv1 = (const float*)d_in[4];
    const float* wo1 = (const float*)d_in[5];
    const float* bq1 = (const float*)d_in[6];
    const float* bk1 = (const float*)d_in[7];
    const float* bv1 = (const float*)d_in[8];
    const float* bo1 = (const float*)d_in[9];
    const float* wq2 = (const float*)d_in[10];
    const float* wk2 = (const float*)d_in[11];
    const float* wv2 = (const float*)d_in[12];
    const float* wo2 = (const float*)d_in[13];
    const float* bq2 = (const float*)d_in[14];
    const float* bk2 = (const float*)d_in[15];
    const float* bv2 = (const float*)d_in[16];
    const float* bo2 = (const float*)d_in[17];
    const float* wff1 = (const float*)d_in[18];
    const float* bff1 = (const float*)d_in[19];
    const float* wff2 = (const float*)d_in[20];
    const float* bff2 = (const float*)d_in[21];
    const float* g1  = (const float*)d_in[22];
    const float* be1 = (const float*)d_in[23];
    const float* g2  = (const float*)d_in[24];
    const float* be2 = (const float*)d_in[25];
    const float* g3  = (const float*)d_in[26];
    const float* be3 = (const float*)d_in[27];

    // Workspace bump allocator
    char* p = (char*)d_ws;
    auto alloc = [&](size_t bytes) -> void* {
        void* r = (void*)p;
        p += (bytes + 255) & ~(size_t)255;
        return r;
    };
    const size_t BSD = (size_t)BSROW * DIM;
    __bf16* xbf    = (__bf16*)alloc(BSD * 2);
    __bf16* encbf  = (__bf16*)alloc(BSD * 2);
    __bf16* wq1t   = (__bf16*)alloc((size_t)DIM * DIM * 2);
    __bf16* wk1t   = (__bf16*)alloc((size_t)DIM * DIM * 2);
    __bf16* wv1t   = (__bf16*)alloc((size_t)DIM * DIM * 2);
    __bf16* wo1t   = (__bf16*)alloc((size_t)DIM * DIM * 2);
    __bf16* wq2t   = (__bf16*)alloc((size_t)DIM * DIM * 2);
    __bf16* wk2t   = (__bf16*)alloc((size_t)DIM * DIM * 2);
    __bf16* wv2t   = (__bf16*)alloc((size_t)DIM * DIM * 2);
    __bf16* wo2t   = (__bf16*)alloc((size_t)DIM * DIM * 2);
    __bf16* wff1t  = (__bf16*)alloc((size_t)DIM * DFF * 2);   // [DFF x DIM]
    __bf16* wff2t  = (__bf16*)alloc((size_t)DIM * DFF * 2);   // [DIM x DFF]
    __bf16* Qb     = (__bf16*)alloc(BSD * 2);
    __bf16* Kbuf   = (__bf16*)alloc(BSD * 2);
    __bf16* Vbuf   = (__bf16*)alloc(BSD * 2);
    __bf16* Vtb    = (__bf16*)alloc(BSD * 2);
    __bf16* attnO  = (__bf16*)alloc(BSD * 2);
    float*  tmpF   = (float*)alloc(BSD * 4);
    float*  o1f    = (float*)alloc(BSD * 4);
    __bf16* o1b    = (__bf16*)alloc(BSD * 2);
    float*  o2f    = (float*)alloc(BSD * 4);
    __bf16* o2b    = (__bf16*)alloc(BSD * 2);
    __bf16* hffb   = (__bf16*)alloc((size_t)BSROW * DFF * 2);
    (void)ws_size; (void)in_sizes; (void)n_in; (void)out_size;

    const int nElem = (int)BSD;
    const int cvtBlocks = nElem / 256;

    // ---- prep: convert activations, transpose+convert weights
    cvt_bf16_kernel<<<cvtBlocks, 256, 0, stream>>>(inputs, xbf, nElem);
    cvt_bf16_kernel<<<cvtBlocks, 256, 0, stream>>>(enc, encbf, nElem);
    const int twDD  = (DIM * DIM) / 256;
    const int twDF  = (DIM * DFF) / 256;
    transpose_w_kernel<<<twDD, 256, 0, stream>>>(wq1, wq1t, DIM, DIM);
    transpose_w_kernel<<<twDD, 256, 0, stream>>>(wk1, wk1t, DIM, DIM);
    transpose_w_kernel<<<twDD, 256, 0, stream>>>(wv1, wv1t, DIM, DIM);
    transpose_w_kernel<<<twDD, 256, 0, stream>>>(wo1, wo1t, DIM, DIM);
    transpose_w_kernel<<<twDD, 256, 0, stream>>>(wq2, wq2t, DIM, DIM);
    transpose_w_kernel<<<twDD, 256, 0, stream>>>(wk2, wk2t, DIM, DIM);
    transpose_w_kernel<<<twDD, 256, 0, stream>>>(wv2, wv2t, DIM, DIM);
    transpose_w_kernel<<<twDD, 256, 0, stream>>>(wo2, wo2t, DIM, DIM);
    transpose_w_kernel<<<twDF, 256, 0, stream>>>(wff1, wff1t, DIM, DFF);
    transpose_w_kernel<<<twDF, 256, 0, stream>>>(wff2, wff2t, DFF, DIM);

    const dim3 attnGrid(SEQ / 64, NH, BSZ);

    // ================= self-attention (causal) =================
    launch_gemm(stream, xbf, DIM, wq1t, bq1, nullptr, nullptr, Qb,   BSROW, DIM, DIM, 0);
    launch_gemm(stream, xbf, DIM, wk1t, bk1, nullptr, nullptr, Kbuf, BSROW, DIM, DIM, 0);
    launch_gemm(stream, xbf, DIM, wv1t, bv1, nullptr, nullptr, Vbuf, BSROW, DIM, DIM, 0);
    transpose_v_kernel<<<cvtBlocks, 256, 0, stream>>>(Vbuf, Vtb);
    attn_kernel<<<attnGrid, 128, 0, stream>>>(Qb, Kbuf, Vtb, attnO, /*causal=*/1);
    launch_gemm(stream, attnO, DIM, wo1t, bo1, inputs, tmpF, nullptr, BSROW, DIM, DIM, 0);
    ln_kernel<<<BSROW, 256, 0, stream>>>(tmpF, g1, be1, o1f, o1b, DIM);

    // ================= cross-attention =================
    launch_gemm(stream, o1b,   DIM, wq2t, bq2, nullptr, nullptr, Qb,   BSROW, DIM, DIM, 0);
    launch_gemm(stream, encbf, DIM, wk2t, bk2, nullptr, nullptr, Kbuf, BSROW, DIM, DIM, 0);
    launch_gemm(stream, encbf, DIM, wv2t, bv2, nullptr, nullptr, Vbuf, BSROW, DIM, DIM, 0);
    transpose_v_kernel<<<cvtBlocks, 256, 0, stream>>>(Vbuf, Vtb);
    attn_kernel<<<attnGrid, 128, 0, stream>>>(Qb, Kbuf, Vtb, attnO, /*causal=*/0);
    launch_gemm(stream, attnO, DIM, wo2t, bo2, o1f, tmpF, nullptr, BSROW, DIM, DIM, 0);
    ln_kernel<<<BSROW, 256, 0, stream>>>(tmpF, g2, be2, o2f, o2b, DIM);

    // ================= FFN =================
    launch_gemm(stream, o2b,  DIM, wff1t, bff1, nullptr, nullptr, hffb, BSROW, DFF, DIM, /*relu=*/1);
    launch_gemm(stream, hffb, DFF, wff2t, bff2, o2f, tmpF, nullptr, BSROW, DIM, DFF, 0);
    ln_kernel<<<BSROW, 256, 0, stream>>>(tmpF, g3, be3, (float*)d_out, nullptr, DIM);
}
